// UCAD_9225589751975
// MI455X (gfx1250) — matrix operand order repaired and checked by reading, not compiled, not run on hardware
//
#include <hip/hip_runtime.h>
#include <hip/hip_bf16.h>
#include <stdint.h>

// ---------------- CDNA5 WMMA types ----------------
typedef __attribute__((ext_vector_type(16))) __bf16 v16bf;
typedef __attribute__((ext_vector_type(8)))  float  v8f;

// ---------------- problem constants ----------------
#define PD      150528      // P*D = 196*768 (flattened feature length)
#define BQ      256         // batch of queries
#define CK      15          // classes (keys)
#define KCHUNK  1536        // floats of K per GEMM wave  (48 WMMA steps of K=32)
#define NCHUNK  98          // PD / KCHUNK
#define NEMB    12544       // embeddings rows
#define DEMB    768         // embedding dim
#define NSAMP   196         // FPS samples
#define FPS_BLOCKS 49

// =====================================================================
// 1) row norms of q (256 rows) and k (15 rows), each of length PD (f32 exact)
// =====================================================================
__global__ void ucad_norms(const float* __restrict__ q, const float* __restrict__ k,
                           float* __restrict__ qn, float* __restrict__ kn) {
    int b = blockIdx.x;
    const float* row;
    float* outp;
    if (b < BQ) { row = q + (size_t)b * PD;        outp = qn + b; }
    else        { row = k + (size_t)(b - BQ) * PD; outp = kn + (b - BQ); }
    float acc = 0.f;
    for (int i = threadIdx.x; i < PD; i += blockDim.x) { float v = row[i]; acc += v * v; }
    __shared__ float red[256];
    red[threadIdx.x] = acc;
    __syncthreads();
    for (int s = 128; s > 0; s >>= 1) {
        if ((int)threadIdx.x < s) red[threadIdx.x] += red[threadIdx.x + s];
        __syncthreads();
    }
    if (threadIdx.x == 0) *outp = red[0];
}

// =====================================================================
// 2) K-split GEMM  S = q @ k^T  via v_wmma_f32_16x16x32_bf16
//    grid = (NCHUNK, 16 M-tiles), one wave (32 threads) per block.
//    Unrolled x2 with two independent accumulators so consecutive WMMAs
//    have no C/D RAW chain. Column 15 of the tile is never consumed by
//    the reduce kernel, so no B-masking is needed (invalid lane reads
//    class 0 -> finite garbage in an unread column).
// =====================================================================
template <int BASE>
__device__ __forceinline__ void pack4(v16bf& v, float4 f) {
    v[BASE + 0] = (__bf16)f.x; v[BASE + 1] = (__bf16)f.y;
    v[BASE + 2] = (__bf16)f.z; v[BASE + 3] = (__bf16)f.w;
}

__device__ __forceinline__ v16bf load_tile_a(const float* qrow, int kk, int hi) {
    // A (16x32 bf16): lane lo = row M, hi selects K sub-ranges:
    //   v[0..7] <- K = 16*hi + 0..7 ;  v[8..15] <- K = 16*hi + 8 + 0..7
    const float4* pa1 = (const float4*)(qrow + kk + 16 * hi);
    const float4* pa2 = (const float4*)(qrow + kk + 16 * hi + 8);
    v16bf a;
    pack4<0>(a, pa1[0]);  pack4<4>(a, pa1[1]);
    pack4<8>(a, pa2[0]);  pack4<12>(a, pa2[1]);
    return a;
}
__device__ __forceinline__ v16bf load_tile_b(const float* krow, int kk, int hi) {
    // B (32x16 bf16): lane lo = col N (class), K range = 16*hi + 0..15
    const float4* pb = (const float4*)(krow + kk + 16 * hi);
    v16bf b;
    pack4<0>(b, pb[0]);  pack4<4>(b, pb[1]);
    pack4<8>(b, pb[2]);  pack4<12>(b, pb[3]);
    return b;
}

__global__ void ucad_gemm_bf16(const float* __restrict__ q, const float* __restrict__ kmat,
                               float* __restrict__ partials) {
    const int lane = threadIdx.x;   // 0..31
    const int hi   = lane >> 4;     // half-wave select
    const int lo   = lane & 15;
    const int chunk = blockIdx.x;        // 0..NCHUNK-1
    const int m0    = blockIdx.y * 16;   // M tile base
    const int kbase = chunk * KCHUNK;

    const float* qrow = q + (size_t)(m0 + lo) * PD + kbase;
    const float* krow = kmat + (size_t)(lo < CK ? lo : 0) * PD + kbase;

    v8f acc0 = {};
    v8f acc1 = {};
    for (int kk = 0; kk < KCHUNK; kk += 64) {
        v16bf a0 = load_tile_a(qrow, kk,      hi);
        v16bf b0 = load_tile_b(krow, kk,      hi);
        v16bf a1 = load_tile_a(qrow, kk + 32, hi);
        v16bf b1 = load_tile_b(krow, kk + 32, hi);
        acc0 = __builtin_amdgcn_wmma_f32_16x16x32_bf16(
            false, a0, false, b0, (short)0, acc0, false, false);
        acc1 = __builtin_amdgcn_wmma_f32_16x16x32_bf16(
            false, a1, false, b1, (short)0, acc1, false, false);
    }
    // C/D layout: VGPR j, lane -> (M = 8*hi + j, N = lo)
    float* outp = partials + (size_t)chunk * (BQ * 16) + (size_t)m0 * 16;
#pragma unroll
    for (int j = 0; j < 8; ++j) {
        int M = 8 * hi + j;
        outp[M * 16 + lo] = acc0[j] + acc1[j];
    }
}

// =====================================================================
// 3) reduce K-chunks, form sqrt distances, min/argmin over 15 classes
//    out[0..255] = min_dist, out[256..511] = argmin (as float)
// =====================================================================
__global__ void ucad_reduce_retrieve(const float* __restrict__ partials,
                                     const float* __restrict__ qn, const float* __restrict__ kn,
                                     float* __restrict__ out) {
    int b = blockIdx.x;
    int c = threadIdx.x;     // 32-thread block, lanes 0..15 active
    __shared__ float sd[16];
    if (c < 16) {
        float s = 0.f;
        const float* p = partials + (size_t)b * 16 + c;
        for (int ch = 0; ch < NCHUNK; ++ch) s += p[(size_t)ch * (BQ * 16)];
        float d = 1e30f;
        if (c < CK) {
            float sq = qn[b] + kn[c] - 2.0f * s;
            d = sqrtf(fmaxf(sq, 1e-12f));
        }
        sd[c] = d;
    }
    __syncthreads();
    if (c == 0) {
        float best = sd[0]; int bi = 0;
        for (int j = 1; j < CK; ++j) if (sd[j] < best) { best = sd[j]; bi = j; }
        out[b]      = best;
        out[BQ + b] = (float)bi;
    }
}

// =====================================================================
// 4) FPS init: |e_n|^2 per row (one wave per row), dists = +inf
// =====================================================================
__global__ void ucad_sqn_init(const float* __restrict__ emb,
                              float* __restrict__ sqn, float* __restrict__ dists) {
    int wid  = (blockIdx.x * blockDim.x + threadIdx.x) >> 5;
    int lane = threadIdx.x & 31;
    if (wid >= NEMB) return;
    const float4* row = (const float4*)(emb + (size_t)wid * DEMB);
    float acc = 0.f;
#pragma unroll
    for (int j = 0; j < DEMB / 128; ++j) {
        float4 v = row[lane + 32 * j];
        acc += v.x * v.x + v.y * v.y + v.z * v.z + v.w * v.w;
    }
    for (int m = 16; m >= 1; m >>= 1) acc += __shfl_xor(acc, m, 32);
    if (lane == 0) { sqn[wid] = acc; dists[wid] = 1e30f; }
}

// =====================================================================
// 5) one FPS step: dist-to-centroid (wave-cooperative dot), running min,
//    device argmax via packed u64 atomicMax (float bits || ~idx :
//    max distance wins; ties prefer the smaller index, like jnp.argmax).
// =====================================================================
__global__ void ucad_fps_step(const float* __restrict__ emb, const float* __restrict__ sqn,
                              float* __restrict__ dists, unsigned long long* __restrict__ slots,
                              int step) {
    const int lane = threadIdx.x & 31;
    const int wInB = threadIdx.x >> 5;
    const int wid  = blockIdx.x * (blockDim.x >> 5) + wInB;
    const int nw   = gridDim.x * (blockDim.x >> 5);

    unsigned int cidx = 0;
    if (step > 0) {
        cidx = ~(unsigned int)(slots[step - 1] & 0xffffffffu);
        if (cidx >= NEMB) cidx = 0;
    }
    const float cc = sqn[cidx];
    const float4* crow = (const float4*)(emb + (size_t)cidx * DEMB);

    unsigned long long best = 0ull;
    for (int n = wid; n < NEMB; n += nw) {
        const float4* row = (const float4*)(emb + (size_t)n * DEMB);
        float acc = 0.f;
#pragma unroll
        for (int j = 0; j < DEMB / 128; ++j) {
            float4 a = row[lane + 32 * j];
            float4 b = crow[lane + 32 * j];
            acc += a.x * b.x + a.y * b.y + a.z * b.z + a.w * b.w;
        }
        for (int m = 16; m >= 1; m >>= 1) acc += __shfl_xor(acc, m, 32);
        float cur  = sqrtf(fmaxf(sqn[n] - 2.0f * acc + cc, 1e-12f));
        float dmin = fminf(dists[n], cur);
        if (lane == 0) dists[n] = dmin;
        unsigned long long pk =
            ((unsigned long long)__float_as_uint(dmin) << 32) | (unsigned int)(~n);
        best = (pk > best) ? pk : best;
    }
    __shared__ unsigned long long wbest[8];
    if (lane == 0) wbest[wInB] = best;
    __syncthreads();
    if (threadIdx.x == 0) {
        unsigned long long bb = wbest[0];
        int nwb = blockDim.x >> 5;
        for (int j = 1; j < nwb; ++j) bb = (wbest[j] > bb) ? wbest[j] : bb;
        atomicMax(&slots[step], bb);
    }
}

// =====================================================================
// 6) gather sampled rows into out[512 ..]
// =====================================================================
__global__ void ucad_gather(const float* __restrict__ emb,
                            const unsigned long long* __restrict__ slots,
                            float* __restrict__ out) {
    int j = blockIdx.x;
    unsigned int idx = 0;
    if (j > 0) {
        idx = ~(unsigned int)(slots[j - 1] & 0xffffffffu);
        if (idx >= NEMB) idx = 0;
    }
    const float* src = emb + (size_t)idx * DEMB;
    float* dst = out + 2 * BQ + (size_t)j * DEMB;
    for (int d = threadIdx.x; d < DEMB; d += blockDim.x) dst[d] = src[d];
}

// =====================================================================
extern "C" void kernel_launch(void* const* d_in, const int* in_sizes, int n_in,
                              void* d_out, int out_size, void* d_ws, size_t ws_size,
                              hipStream_t stream) {
    const float* q   = (const float*)d_in[0];   // [256, 196, 768]
    const float* k   = (const float*)d_in[1];   // [15, 196, 768]
    const float* emb = (const float*)d_in[2];   // [12544, 768]
    (void)in_sizes; (void)n_in; (void)out_size; (void)ws_size;
    float* out = (float*)d_out;

    char* ws = (char*)d_ws;
    size_t off = 0;
    float* partials = (float*)(ws + off); off += (size_t)NCHUNK * BQ * 16 * sizeof(float);
    float* qn       = (float*)(ws + off); off += BQ * sizeof(float);
    float* kn       = (float*)(ws + off); off += 16 * sizeof(float);
    float* sqn      = (float*)(ws + off); off += NEMB * sizeof(float);
    float* dists    = (float*)(ws + off); off += NEMB * sizeof(float);
    off = (off + 7) & ~(size_t)7;
    unsigned long long* slots = (unsigned long long*)(ws + off); off += NSAMP * 8;

    // clear argmax slots every call (graph-capturable async memset)
    hipMemsetAsync(slots, 0, NSAMP * 8, stream);

    // --- retrieve_key ---
    ucad_norms<<<BQ + CK, 256, 0, stream>>>(q, k, qn, kn);
    dim3 ggrid(NCHUNK, 16);
    ucad_gemm_bf16<<<ggrid, 32, 0, stream>>>(q, k, partials);
    ucad_reduce_retrieve<<<BQ, 32, 0, stream>>>(partials, qn, kn, out);

    // --- furthest point sampling ---
    ucad_sqn_init<<<(NEMB * 32) / 256, 256, 0, stream>>>(emb, sqn, dists);
    for (int i = 0; i < NSAMP - 1; ++i)
        ucad_fps_step<<<FPS_BLOCKS, 256, 0, stream>>>(emb, sqn, dists, slots, i);
    ucad_gather<<<NSAMP, 256, 0, stream>>>(emb, slots, out);
}
